// GPT_78331613544707
// MI455X (gfx1250) — matrix operand order repaired and checked by reading, not compiled, not run on hardware
//
#include <hip/hip_runtime.h>
#include <hip/hip_bf16.h>
#include <math.h>

// ---------------------------------------------------------------------------
// GPT forward on gfx1250 (MI455X). All GEMMs via v_wmma_f32_16x16x32_f16.
// Block = 256 threads = 8 waves (2x4); wave tile 32x64 (2x4 WMMA accumulators);
// block tile 64x256; K step 32; double-buffered LDS with register staging.
// ---------------------------------------------------------------------------

typedef __attribute__((ext_vector_type(16))) _Float16 v16h;
typedef __attribute__((ext_vector_type(8)))  _Float16 h8;
typedef __attribute__((ext_vector_type(8)))  float    v8f;

union Frag16 { v16h v; h8 h[2]; };

#define BN 64      // rows (token dim) per block
#define BM 256     // cols per block
#define BK 32      // K per WMMA step
#define LDSP 40    // padded row stride (f16 units) to dodge bank conflicts

// ---------------------------------------------------------------------------
// Embedding: x[n,c] = tok_emb[idx[n],c] + pos_emb[n%T, c]
// ---------------------------------------------------------------------------
__global__ void embed_k(const int* __restrict__ idx, const float* __restrict__ tok,
                        const float* __restrict__ pos, float* __restrict__ x,
                        int T, int Cn) {
    int n = blockIdx.x;
    int t = n % T;
    int token = idx[n];
    const float* te = tok + (size_t)token * Cn;
    const float* pe = pos + (size_t)t * Cn;
    float* o = x + (size_t)n * Cn;
    for (int c = threadIdx.x; c < Cn; c += blockDim.x) o[c] = te[c] + pe[c];
}

// ---------------------------------------------------------------------------
// LayerNorm: one block per row
// ---------------------------------------------------------------------------
__global__ __launch_bounds__(256) void layernorm_k(const float* __restrict__ in,
                                                   const float* __restrict__ g,
                                                   const float* __restrict__ b,
                                                   float* __restrict__ out, int Cn) {
    int row = blockIdx.x;
    const float* p = in + (size_t)row * Cn;
    __shared__ float r1[256], r2[256];
    int tid = threadIdx.x;
    float s = 0.f, s2 = 0.f;
    for (int c = tid; c < Cn; c += 256) { float v = p[c]; s += v; s2 += v * v; }
    r1[tid] = s; r2[tid] = s2; __syncthreads();
    for (int st = 128; st > 0; st >>= 1) {
        if (tid < st) { r1[tid] += r1[tid + st]; r2[tid] += r2[tid + st]; }
        __syncthreads();
    }
    float mu  = r1[0] / (float)Cn;
    float var = r2[0] / (float)Cn - mu * mu;
    float inv = rsqrtf(var + 1e-5f);
    for (int c = tid; c < Cn; c += 256)
        out[(size_t)row * Cn + c] = (p[c] - mu) * inv * g[c] + b[c];
}

// ---------------------------------------------------------------------------
// Causal softmax over att rows (in place). One block per (b,h,q) row.
// ---------------------------------------------------------------------------
__global__ __launch_bounds__(256) void softmax_causal_k(float* __restrict__ att,
                                                        int T, float scale) {
    int row = blockIdx.x;          // 0 .. B*H*T-1
    int q = row % T;
    float* p = att + (size_t)row * T;
    __shared__ float red[256];
    int tid = threadIdx.x;
    float mx = -1e30f;
    for (int k = tid; k < T; k += 256)
        if (k <= q) mx = fmaxf(mx, p[k] * scale);
    red[tid] = mx; __syncthreads();
    for (int s = 128; s > 0; s >>= 1) {
        if (tid < s) red[tid] = fmaxf(red[tid], red[tid + s]);
        __syncthreads();
    }
    mx = red[0]; __syncthreads();
    float sum = 0.f;
    for (int k = tid; k < T; k += 256)
        if (k <= q) sum += expf(p[k] * scale - mx);
    red[tid] = sum; __syncthreads();
    for (int s = 128; s > 0; s >>= 1) {
        if (tid < s) red[tid] += red[tid + s];
        __syncthreads();
    }
    float inv = 1.0f / red[0];
    for (int k = tid; k < T; k += 256)
        p[k] = (k <= q) ? expf(p[k] * scale - mx) * inv : 0.0f;
}

// ---------------------------------------------------------------------------
// Batched WMMA GEMM.
//   TRANSB=true : C[n,m] = sum_k A[n,k] * Bm[m,k]   (x @ w.T — the common case)
//   TRANSB=false: C[n,m] = sum_k A[n,k] * Bm[k,m]   (att @ V)
// EPI: 0=none, 1=add Res[n,m], 2=exact GELU.
// MG : guard m against M (only needed when M % BM != 0, i.e. att @ V).
// Batch z: offsets = (z/Hdiv)*s?b + (z%Hdiv)*s?h.
// ---------------------------------------------------------------------------
template<int EPI, bool TRANSB, bool MG>
__global__ __launch_bounds__(256) void gemm_wmma(
        const float* __restrict__ A, const float* __restrict__ Bm,
        const float* Res, float* C,
        int N, int M, int K, int lda, int ldb, int ldc,
        int Hdiv, long long sAb, long long sAh, long long sBb, long long sBh,
        long long sCb, long long sCh)
{
    __shared__ _Float16 sA[2][BN * LDSP];
    __shared__ _Float16 sB[2][BM * LDSP];

    const int t = threadIdx.x;
    const int z = blockIdx.z;
    const int zb = z / Hdiv, zh = z % Hdiv;
    const size_t aOff = (size_t)zb * sAb + (size_t)zh * sAh;
    const size_t bOff = (size_t)zb * sBb + (size_t)zh * sBh;
    const size_t cOff = (size_t)zb * sCb + (size_t)zh * sCh;
    const int nBase = blockIdx.y * BN;
    const int mBase = blockIdx.x * BM;

    const int wave = t >> 5, lane = t & 31;
    const int wr = wave >> 2, wc = wave & 3;       // 2x4 wave grid
    const int lrow = lane & 15, hlf = lane >> 4;   // ISA 7.12.2 lane mapping

    v8f acc[2][4];
    {
        v8f vz = {0.f,0.f,0.f,0.f,0.f,0.f,0.f,0.f};
        #pragma unroll
        for (int i = 0; i < 2; ++i)
            #pragma unroll
            for (int j = 0; j < 4; ++j) acc[i][j] = vz;
    }

    // loader lane roles
    const int arow = t >> 2;            // 0..63
    const int akc  = (t & 3) << 3;      // 0,8,16,24
    const int brow = t;                 // TRANSB: one full B row per thread
    const int nkk  = t >> 3;            // NN: 0..31
    const int nmo  = (t & 7) << 5;      // NN: 0,32,..,224

    // register staging for the next tile
    h8 ra;                              // A: 8 halves
    h8 rb[4];                           // B (TRANSB): 32 halves
    float rbn[32];                      // B (NN): 32 floats

    auto fetch = [&](int k0) {
        // ---- A: 64x32, 8 contiguous floats per thread
        const float* srcA = A + aOff + (size_t)(nBase + arow) * lda + (k0 + akc);
        float4 f0 = *(const float4*)(srcA);
        float4 f1 = *(const float4*)(srcA + 4);
        ra[0]=(_Float16)f0.x; ra[1]=(_Float16)f0.y; ra[2]=(_Float16)f0.z; ra[3]=(_Float16)f0.w;
        ra[4]=(_Float16)f1.x; ra[5]=(_Float16)f1.y; ra[6]=(_Float16)f1.z; ra[7]=(_Float16)f1.w;
        if (TRANSB) {
            // 256x32: one row (32 contiguous floats) per thread; M % BM == 0 here.
            const float* srcB = Bm + bOff + (size_t)(mBase + brow) * ldb + k0;
            #pragma unroll
            for (int u = 0; u < 4; ++u) {
                float4 g0 = *(const float4*)(srcB + u * 8);
                float4 g1 = *(const float4*)(srcB + u * 8 + 4);
                rb[u][0]=(_Float16)g0.x; rb[u][1]=(_Float16)g0.y;
                rb[u][2]=(_Float16)g0.z; rb[u][3]=(_Float16)g0.w;
                rb[u][4]=(_Float16)g1.x; rb[u][5]=(_Float16)g1.y;
                rb[u][6]=(_Float16)g1.z; rb[u][7]=(_Float16)g1.w;
            }
        } else {
            // B is [K,M]; gather a k-row slice, transposing into lds later.
            const float* srcB = Bm + bOff + (size_t)(k0 + nkk) * ldb + (mBase + nmo);
            #pragma unroll
            for (int u = 0; u < 32; ++u) {
                int gm = mBase + nmo + u;
                rbn[u] = (!MG || gm < M) ? srcB[u] : 0.0f;
            }
        }
    };

    auto stage = [&](int buf) {
        *(h8*)(&sA[buf][arow * LDSP + akc]) = ra;
        if (TRANSB) {
            #pragma unroll
            for (int u = 0; u < 4; ++u)
                *(h8*)(&sB[buf][brow * LDSP + u * 8]) = rb[u];
        } else {
            #pragma unroll
            for (int u = 0; u < 32; ++u)
                sB[buf][(nmo + u) * LDSP + nkk] = (_Float16)rbn[u];
        }
    };

    auto compute = [&](int buf) {
        Frag16 fa[2], fb[4];
        #pragma unroll
        for (int i = 0; i < 2; ++i) {
            const _Float16* pa = &sA[buf][(wr * 32 + i * 16 + lrow) * LDSP + hlf * 8];
            fa[i].h[0] = *(const h8*)pa;          // K = hlf*8 .. +7
            fa[i].h[1] = *(const h8*)(pa + 16);   // K = 16 + hlf*8 .. +7
        }
        #pragma unroll
        for (int j = 0; j < 4; ++j) {
            const _Float16* pb = &sB[buf][(wc * 64 + j * 16 + lrow) * LDSP + hlf * 8];
            fb[j].h[0] = *(const h8*)pb;          // K = hlf*8 .. +7
            fb[j].h[1] = *(const h8*)(pb + 16);   // K = 16 + hlf*8 .. +7
        }
        #pragma unroll
        for (int i = 0; i < 2; ++i)
            #pragma unroll
            for (int j = 0; j < 4; ++j)
                acc[i][j] = __builtin_amdgcn_wmma_f32_16x16x32_f16(
                    false, fa[i].v, false, fb[j].v, (short)0, acc[i][j], false, false);
    };

    // ---- software-pipelined main loop (double-buffered LDS)
    const int ktiles = K / BK;
    fetch(0);
    stage(0);
    __syncthreads();
    for (int kt = 0; kt < ktiles; ++kt) {
        const int cur = kt & 1;
        if (kt + 1 < ktiles) fetch((kt + 1) * BK);   // global loads overlap WMMA
        compute(cur);
        if (kt + 1 < ktiles) stage(1 - cur);
        __syncthreads();
    }

    // ---- epilogue + store (C layout: VGPR r -> M=r+8*hlf, N=lrow)
    #pragma unroll
    for (int i = 0; i < 2; ++i) {
        #pragma unroll
        for (int j = 0; j < 4; ++j) {
            int m = mBase + wc * 64 + j * 16 + lrow;
            if (MG && m >= M) continue;
            #pragma unroll
            for (int r = 0; r < 8; ++r) {
                int n = nBase + wr * 32 + i * 16 + hlf * 8 + r;
                float v = acc[i][j][r];
                if (EPI == 1) v += Res[cOff + (size_t)n * ldc + m];
                if (EPI == 2) v = 0.5f * v * (1.0f + erff(v * 0.70710678118f));
                C[cOff + (size_t)n * ldc + m] = v;
            }
        }
    }
}

// ---------------------------------------------------------------------------
extern "C" void kernel_launch(void* const* d_in, const int* in_sizes, int n_in,
                              void* d_out, int out_size, void* d_ws, size_t ws_size,
                              hipStream_t stream) {
    (void)in_sizes; (void)n_in; (void)out_size; (void)ws_size;
    const int*   idx    = (const int*)  d_in[0];
    const float* tok    = (const float*)d_in[1];
    const float* pos    = (const float*)d_in[2];
    const float* ln1_g  = (const float*)d_in[3];
    const float* ln1_b  = (const float*)d_in[4];
    const float* attn_w = (const float*)d_in[5];
    const float* proj_w = (const float*)d_in[6];
    const float* ln2_g  = (const float*)d_in[7];
    const float* ln2_b  = (const float*)d_in[8];
    const float* mlp_w1 = (const float*)d_in[9];
    const float* mlp_w2 = (const float*)d_in[10];
    const float* lnf_g  = (const float*)d_in[11];
    const float* lnf_b  = (const float*)d_in[12];
    const float* head_w = (const float*)d_in[13];

    const int B = 4, T = 512, Cc = 1024, L = 8, V = 32000, H = 16;
    const int NT = B * T;                    // 2048 tokens

    float* ws  = (float*)d_ws;
    float* x   = ws;                                  // [NT, C]
    float* h   = x   + (size_t)NT * Cc;               // [NT, C]  (LN outputs)
    float* qkv = h   + (size_t)NT * Cc;               // [NT, 3C]
    float* att = qkv + (size_t)NT * 3 * Cc;           // [B,H,T,T]
    float* y   = att + (size_t)B * H * T * T;         // [NT, C]
    float* a   = y   + (size_t)NT * Cc;               // [NT, C]
    float* hid = a   + (size_t)NT * Cc;               // [NT, 4C]

    embed_k<<<NT, 256, 0, stream>>>(idx, tok, pos, x, T, Cc);

    const long long sQKVb = (long long)T * 3 * Cc;    // per-batch stride in qkv
    const long long sAttB = (long long)H * T * T;     // per-b stride in att
    const long long sAttH = (long long)T * T;         // per-h stride in att

    for (int l = 0; l < L; ++l) {
        // h = LN1(x)
        layernorm_k<<<NT, 256, 0, stream>>>(x, ln1_g + (size_t)l * Cc, ln1_b + (size_t)l * Cc, h, Cc);
        // qkv = h @ attn_w.T   [2048 x 3072]
        {
            dim3 g((3 * Cc + BM - 1) / BM, NT / BN, 1);
            gemm_wmma<0, true, false><<<g, 256, 0, stream>>>(h, attn_w + (size_t)l * 3 * Cc * Cc,
                nullptr, qkv, NT, 3 * Cc, Cc, Cc, Cc, 3 * Cc, 1, 0, 0, 0, 0, 0, 0);
        }
        // att = Q @ K.T per (b,h)  [512 x 512], K-dim = 64
        {
            dim3 g((T + BM - 1) / BM, T / BN, B * H);
            gemm_wmma<0, true, false><<<g, 256, 0, stream>>>(qkv, qkv + Cc, nullptr, att,
                T, T, 64, 3 * Cc, 3 * Cc, T, H,
                sQKVb, 64, sQKVb, 64, sAttB, sAttH);
        }
        softmax_causal_k<<<B * H * T, 256, 0, stream>>>(att, T, 0.125f);
        // y = att @ V per (b,h)  [512 x 64], K-dim = 512  (NN, M-guarded)
        {
            dim3 g((64 + BM - 1) / BM, T / BN, B * H);
            gemm_wmma<0, false, true><<<g, 256, 0, stream>>>(att, qkv + 2 * Cc, nullptr, y,
                T, 64, T, T, 3 * Cc, Cc, H,
                sAttB, sAttH, sQKVb, 64, (long long)T * Cc, 64);
        }
        // a = x + y @ proj_w.T
        {
            dim3 g((Cc + BM - 1) / BM, NT / BN, 1);
            gemm_wmma<1, true, false><<<g, 256, 0, stream>>>(y, proj_w + (size_t)l * Cc * Cc,
                x, a, NT, Cc, Cc, Cc, Cc, Cc, 1, 0, 0, 0, 0, 0, 0);
        }
        // h = LN2(a)
        layernorm_k<<<NT, 256, 0, stream>>>(a, ln2_g + (size_t)l * Cc, ln2_b + (size_t)l * Cc, h, Cc);
        // hid = gelu(h @ w1.T)   [2048 x 4096]
        {
            dim3 g((4 * Cc + BM - 1) / BM, NT / BN, 1);
            gemm_wmma<2, true, false><<<g, 256, 0, stream>>>(h, mlp_w1 + (size_t)l * 4 * Cc * Cc,
                nullptr, hid, NT, 4 * Cc, Cc, Cc, Cc, 4 * Cc, 1, 0, 0, 0, 0, 0, 0);
        }
        // x = x + hid @ w2.T     (outer residual adds x, not a)
        {
            dim3 g((Cc + BM - 1) / BM, NT / BN, 1);
            gemm_wmma<1, true, false><<<g, 256, 0, stream>>>(hid, mlp_w2 + (size_t)l * Cc * 4 * Cc,
                x, x, NT, Cc, 4 * Cc, 4 * Cc, 4 * Cc, Cc, 1, 0, 0, 0, 0, 0, 0);
        }
    }

    // final LN + LM head
    layernorm_k<<<NT, 256, 0, stream>>>(x, lnf_g, lnf_b, h, Cc);
    {
        dim3 g((V + BM - 1) / BM, NT / BN, 1);
        gemm_wmma<0, true, false><<<g, 256, 0, stream>>>(h, head_w, nullptr, (float*)d_out,
            NT, V, Cc, Cc, Cc, V, 1, 0, 0, 0, 0, 0, 0);
    }
}
